// MultivariateGraphAttnDetector_19713899888817
// MI455X (gfx1250) — compile-verified, matrix operand order checked by reading, not run on hardware
//
#include <hip/hip_runtime.h>
#include <hip/hip_bf16.h>

#define DEV static __device__ __forceinline__

typedef __bf16 bf16;
typedef __attribute__((ext_vector_type(16))) __bf16 v16bf;
typedef __attribute__((ext_vector_type(8)))  __bf16 v8bf;
typedef __attribute__((ext_vector_type(8)))  float  v8f;

namespace {
constexpr int B  = 16;
constexpr int T  = 4098;
constexpr int D  = 256;
constexpr int W  = 2048;
constexpr int C3 = 768;   // 3*D, GRU input width and 3*H gate width
}

// ---------------------------------------------------------------------------
// WMMA helpers (wave32, 16x16x32 bf16, f32 accum)
// A fragment and (n-major-staged) B fragment share one layout:
//   lane 0-15 : row = lane,    k = {0..7, 16..23}
//   lane16-31 : row = lane-16, k = {8..15, 24..31}
// k runs contiguously in groups of 8 -> two 16B vector loads per fragment.
// ---------------------------------------------------------------------------
DEV v16bf load_frag(const bf16* tile, int ld) {
  const int lane = threadIdx.x & 31;
  const bf16* p = tile + (size_t)(lane & 15) * ld + ((lane >> 4) << 3);
  v8bf lo = *(const v8bf*)(p);
  v8bf hi = *(const v8bf*)(p + 16);
  v16bf r;
#pragma unroll
  for (int e = 0; e < 8; ++e) { r[e] = lo[e]; r[8 + e] = hi[e]; }
  return r;
}

DEV v8f wmma_bf16(v16bf a, v16bf b, v8f c) {
  return __builtin_amdgcn_wmma_f32_16x16x32_bf16(false, a, false, b, (short)0, c,
                                                 false, false);
}

// C/D layout: VGPR r -> M = r (lanes 0-15) or 8+r (lanes 16-31); N = lane&15
DEV void store_c_f32(float* out, int ld, v8f c) {
  const int lane = threadIdx.x & 31;
  const int n = lane & 15, mb = (lane >> 4) << 3;
#pragma unroll
  for (int r = 0; r < 8; ++r) out[(size_t)(mb + r) * ld + n] = c[r];
}
DEV void store_c_bf16(bf16* out, int ld, v8f c) {
  const int lane = threadIdx.x & 31;
  const int n = lane & 15, mb = (lane >> 4) << 3;
#pragma unroll
  for (int r = 0; r < 8; ++r) out[(size_t)(mb + r) * ld + n] = (bf16)c[r];
}

// ---------------------------------------------------------------------------
// block reductions (256 threads = 8 waves)
// ---------------------------------------------------------------------------
DEV float wave_sum(float v) {
#pragma unroll
  for (int o = 16; o > 0; o >>= 1) v += __shfl_down(v, o, 32);
  return v;
}
DEV float wave_max(float v) {
#pragma unroll
  for (int o = 16; o > 0; o >>= 1) v = fmaxf(v, __shfl_down(v, o, 32));
  return v;
}
DEV float block_sum256(float v, float* sbuf) {
  v = wave_sum(v);
  if ((threadIdx.x & 31) == 0) sbuf[threadIdx.x >> 5] = v;
  __syncthreads();
  if (threadIdx.x == 0) {
    float t = 0.f;
#pragma unroll
    for (int i = 0; i < 8; ++i) t += sbuf[i];
    sbuf[0] = t;
  }
  __syncthreads();
  return sbuf[0];
}
DEV float block_max256(float v, float* sbuf) {
  v = wave_max(v);
  if ((threadIdx.x & 31) == 0) sbuf[threadIdx.x >> 5] = v;
  __syncthreads();
  if (threadIdx.x == 0) {
    float t = sbuf[0];
#pragma unroll
    for (int i = 1; i < 8; ++i) t = fmaxf(t, sbuf[i]);
    sbuf[0] = t;
  }
  __syncthreads();
  return sbuf[0];
}

DEV float lrelu(float s) { return s >= 0.f ? s : 0.2f * s; }

// ---------------------------------------------------------------------------
// K1: moving average (K=3, VALID) + pairwise max pool -> pooled (B,D,W) f32
//     also writes xp into cat[:, :, 0:256] (bf16).   thread = d, block = (b,16 w)
// ---------------------------------------------------------------------------
__global__ __launch_bounds__(256) void prep_kernel(const float* __restrict__ x,
                                                   float* __restrict__ pooled,
                                                   bf16* __restrict__ cat) {
  const int d = threadIdx.x;
  const int b = blockIdx.x >> 7;          // W/16 = 128 chunks
  const int w0 = (blockIdx.x & 127) << 4;
  const float inv3 = 1.0f / 3.0f;
  const float* xb = x + (size_t)b * T * D + d;
  float vals[16];
  float r0 = xb[(size_t)(2 * w0 + 0) * D];
  float r1 = xb[(size_t)(2 * w0 + 1) * D];
  float r2 = xb[(size_t)(2 * w0 + 2) * D];
#pragma unroll
  for (int i = 0; i < 16; ++i) {
    float r3 = xb[(size_t)(2 * w0 + 2 * i + 3) * D];
    int t4 = 2 * w0 + 2 * i + 4; if (t4 > T - 1) t4 = T - 1;  // last value unused
    float r4 = xb[(size_t)t4 * D];
    float a0 = (r0 + r1 + r2) * inv3;
    float a1 = (r1 + r2 + r3) * inv3;
    vals[i] = fmaxf(a0, a1);
    r0 = r2; r1 = r3; r2 = r4;
  }
  float* prow = pooled + ((size_t)b * D + d) * W + w0;
#pragma unroll
  for (int i = 0; i < 16; ++i) prow[i] = vals[i];
  bf16* cb = cat + ((size_t)b * W + w0) * C3 + d;
#pragma unroll
  for (int i = 0; i < 16; ++i) cb[(size_t)i * C3] = (bf16)vals[i];
}

// ---------------------------------------------------------------------------
// K2: time-axis GAT node scalars  k_t[b,i], q_t[b,i] = pooled[b,:,i] . wk/wq
// grid = B*W/256 ; consecutive threads hit consecutive i -> coalesced
// ---------------------------------------------------------------------------
__global__ __launch_bounds__(256) void time_kq_kernel(const float* __restrict__ pooled,
                                                      const float* __restrict__ wk,
                                                      const float* __restrict__ wq,
                                                      float* __restrict__ kt,
                                                      float* __restrict__ qt) {
  const int idx = blockIdx.x * 256 + threadIdx.x;   // b*W + i
  const int b = idx >> 11, i = idx & 2047;
  const float* p = pooled + (size_t)b * D * W + i;
  float sk = 0.f, sq = 0.f;
  for (int d = 0; d < D; ++d) {
    float v = p[(size_t)d * W];
    sk += v * wk[d]; sq += v * wq[d];
  }
  kt[idx] = sk; qt[idx] = sq;
}

// K2b: feature-axis scalars, grid = B*D/256
__global__ __launch_bounds__(256) void feat_kq_kernel(const float* __restrict__ pooled,
                                                      const float* __restrict__ wk,
                                                      const float* __restrict__ wq,
                                                      float* __restrict__ kf,
                                                      float* __restrict__ qf) {
  const int idx = blockIdx.x * 256 + threadIdx.x;   // b*D + i
  const int b = idx >> 8, i = idx & 255;
  const float* p = pooled + ((size_t)b * D + i) * W;
  float sk = 0.f, sq = 0.f;
  for (int w = 0; w < W; ++w) { float v = p[w]; sk += v * wk[w]; sq += v * wq[w]; }
  kf[idx] = sk; qf[idx] = sq;
}

// K3: per-batch max of q (lrelu is monotone -> rowmax_i = lrelu(k_i + qmax_b))
__global__ __launch_bounds__(256) void colmax_kernel(const float* __restrict__ q,
                                                     int n, float* __restrict__ qmax) {
  __shared__ float sbuf[8];
  const float* qb = q + (size_t)blockIdx.x * n;
  float m = -3.0e38f;
  for (int j = threadIdx.x; j < n; j += 256) m = fmaxf(m, qb[j]);
  m = block_max256(m, sbuf);
  if (threadIdx.x == 0) qmax[blockIdx.x] = m;
}

// K4: per-row softmax stats: rmax[b,i], rscale[b,i] = 1/sum_j exp(lrelu(k+q)-rmax)
__global__ __launch_bounds__(256) void rowstats_kernel(const float* __restrict__ kv,
                                                       const float* __restrict__ qv,
                                                       const float* __restrict__ qmax,
                                                       int nrows, int ncols,
                                                       float* __restrict__ rmax,
                                                       float* __restrict__ rscale) {
  __shared__ float sbuf[8];
  const int brow = blockIdx.x;
  const int b = brow / nrows;
  const float ki = kv[brow];
  const float rm = lrelu(ki + qmax[b]);
  const float* qb = qv + (size_t)b * ncols;
  float s = 0.f;
  for (int j = threadIdx.x; j < ncols; j += 256)
    s += __expf(lrelu(ki + qb[j]) - rm);
  s = block_sum256(s, sbuf);
  if (threadIdx.x == 0) { rmax[brow] = rm; rscale[brow] = 1.0f / s; }
}

// ---------------------------------------------------------------------------
// K5: time-GAT WMMA GEMM, attention weights generated on the fly (never
// materialize the 2048x2048 matrix). Block = 8 waves: 16 rows (i) x 256 cols (d),
// K loop over j in steps of 32. B tile staged n-major in LDS.
// Writes h_time -> cat[:, :, 512:768]
// ---------------------------------------------------------------------------
__global__ __launch_bounds__(256) void gat_time_gemm(const float* __restrict__ pooled,
                                                     const float* __restrict__ kt,
                                                     const float* __restrict__ qt,
                                                     const float* __restrict__ rmax,
                                                     const float* __restrict__ rscale,
                                                     bf16* __restrict__ cat) {
  __shared__ __align__(16) bf16 sA[16 * 32];
  __shared__ __align__(16) bf16 sB[256 * 32];   // n-major: sB[d][k]
  __shared__ float sRow[48];                    // k, rmax, rscale per row
  const int b  = blockIdx.x >> 7;
  const int i0 = (blockIdx.x & 127) << 4;
  const int tid = threadIdx.x;
  const int n0 = (tid >> 5) << 5;               // wave's 32-column group
  const float* pb = pooled + (size_t)b * D * W;
  const float* qb = qt + ((size_t)b << 11);
  if (tid < 16) {
    int r = (b << 11) + i0 + tid;
    sRow[tid] = kt[r]; sRow[16 + tid] = rmax[r]; sRow[32 + tid] = rscale[r];
  }
  v8f acc0 = {}, acc1 = {};
  for (int j0 = 0; j0 < W; j0 += 32) {
    __syncthreads();
    // A[m][k] = softmax weight a(i0+m, j0+k), built in-register
    {
      int e = tid * 2;
#pragma unroll
      for (int u = 0; u < 2; ++u, ++e) {
        int m = e >> 5, k = e & 31;
        float s = lrelu(sRow[m] + qb[j0 + k]);
        sA[m * 32 + k] = (bf16)(__expf(s - sRow[16 + m]) * sRow[32 + m]);
      }
    }
    // B[n=d][k=j] = pooled[b][d][j0+k] ; one 32-wide row per thread
    {
      const float* pr = pb + (size_t)tid * W + j0;
#pragma unroll
      for (int k = 0; k < 32; ++k) sB[tid * 32 + k] = (bf16)pr[k];
    }
    __syncthreads();
    v16bf a = load_frag(sA, 32);
    acc0 = wmma_bf16(a, load_frag(sB + (size_t)n0 * 32, 32), acc0);
    acc1 = wmma_bf16(a, load_frag(sB + (size_t)(n0 + 16) * 32, 32), acc1);
  }
  bf16* out = cat + ((size_t)b * W + i0) * C3 + 512 + n0;
  store_c_bf16(out, C3, acc0);
  store_c_bf16(out + 16, C3, acc1);
}

// ---------------------------------------------------------------------------
// K6: feature attention matrix (it IS an output) + bf16 copy for the GEMM
// ---------------------------------------------------------------------------
__global__ __launch_bounds__(256) void feat_attn_kernel(const float* __restrict__ kf,
                                                        const float* __restrict__ qf,
                                                        const float* __restrict__ qmaxf,
                                                        float* __restrict__ attn_out,
                                                        bf16* __restrict__ attn_bf) {
  __shared__ float sbuf[8];
  const int brow = blockIdx.x;          // b*D + i
  const int b = brow >> 8;
  const float ki = kf[brow];
  const float rm = lrelu(ki + qmaxf[b]);
  const int j = threadIdx.x;
  float e = __expf(lrelu(ki + qf[(b << 8) + j]) - rm);
  float s = block_sum256(e, sbuf);
  float a = e / s;
  attn_out[(size_t)brow * D + j] = a;
  attn_bf[(size_t)brow * D + j] = (bf16)a;
}

// K7: feature-GAT WMMA GEMM: h_feat[b,w,i] = sum_j a_f[b,i,j] * pooled[b,j,w]
// Block: 16 w-rows x 256 i-cols, K over j.  Writes cat[:, :, 256:512]
__global__ __launch_bounds__(256) void gat_feat_gemm(const float* __restrict__ pooled,
                                                     const bf16* __restrict__ attn_bf,
                                                     bf16* __restrict__ cat) {
  __shared__ __align__(16) bf16 sA[16 * 32];
  __shared__ __align__(16) bf16 sB[256 * 32];
  const int b  = blockIdx.x >> 7;
  const int w0 = (blockIdx.x & 127) << 4;
  const int tid = threadIdx.x;
  const int n0 = (tid >> 5) << 5;
  const float* pb = pooled + (size_t)b * D * W;
  const bf16* ab = attn_bf + (size_t)b * D * D;
  v8f acc0 = {}, acc1 = {};
  for (int j0 = 0; j0 < D; j0 += 32) {
    __syncthreads();
    {
      int e = tid * 2;
#pragma unroll
      for (int u = 0; u < 2; ++u, ++e) {
        int m = e >> 5, k = e & 31;
        sA[m * 32 + k] = (bf16)pb[(size_t)(j0 + k) * W + w0 + m];
      }
    }
    {
      const bf16* ar = ab + (size_t)tid * D + j0;   // n = i = tid
#pragma unroll
      for (int k = 0; k < 32; ++k) sB[tid * 32 + k] = ar[k];
    }
    __syncthreads();
    v16bf a = load_frag(sA, 32);
    acc0 = wmma_bf16(a, load_frag(sB + (size_t)n0 * 32, 32), acc0);
    acc1 = wmma_bf16(a, load_frag(sB + (size_t)(n0 + 16) * 32, 32), acc1);
  }
  bf16* out = cat + ((size_t)b * W + w0) * C3 + 256 + n0;
  store_c_bf16(out, C3, acc0);
  store_c_bf16(out + 16, C3, acc1);
}

// K8: f32 -> bf16 (whh is (768,256) row-major == n-major B layout already)
__global__ __launch_bounds__(256) void f32_to_bf16_kernel(const float* __restrict__ s,
                                                          bf16* __restrict__ d, int n) {
  int i = blockIdx.x * 256 + threadIdx.x;
  if (i < n) d[i] = (bf16)s[i];
}

// ---------------------------------------------------------------------------
// K9: GRU input projection as one big WMMA GEMM:
//   gi[w,b,:] = cat[b,w,:] @ wih^T + bih       M=32768, N=768, K=768
// Block: 8 waves = 2(M) x 4(N) tiles of 16x16 -> 32x64 tile, 24 K steps.
// ---------------------------------------------------------------------------
__global__ __launch_bounds__(256) void gru_in_gemm(const bf16* __restrict__ cat,
                                                   const float* __restrict__ wih,
                                                   const float* __restrict__ bih,
                                                   float* __restrict__ gi) {
  __shared__ __align__(16) bf16 sA[32 * 32];
  __shared__ __align__(16) bf16 sB[64 * 32];
  const int bx = blockIdx.x;
  const int r0 = (bx / 12) * 32;          // row base (rows are b*W + w)
  const int nblk = (bx % 12) * 64;
  const int tid = threadIdx.x;
  const int wv = tid >> 5;
  const int m0 = (wv & 1) << 4;
  const int n0 = (wv >> 1) << 4;
  v8f acc = {};
  for (int k0 = 0; k0 < C3; k0 += 32) {
    __syncthreads();
    {   // A: 32x32 from cat (bf16, contiguous k)
      int m = tid >> 3, kk = (tid & 7) * 4;
      const bf16* src = cat + (size_t)(r0 + m) * C3 + k0 + kk;
#pragma unroll
      for (int u = 0; u < 4; ++u) sA[m * 32 + kk + u] = src[u];
    }
    {   // B n-major: sB[n][k] = wih[(nblk+n)*768 + k0+k], cvt f32->bf16
      int n = tid >> 2, kk = (tid & 3) * 8;
      const float* src = wih + (size_t)(nblk + n) * C3 + k0 + kk;
#pragma unroll
      for (int u = 0; u < 8; ++u) sB[n * 32 + kk + u] = (bf16)src[u];
    }
    __syncthreads();
    acc = wmma_bf16(load_frag(sA + (size_t)m0 * 32, 32),
                    load_frag(sB + (size_t)n0 * 32, 32), acc);
  }
  // epilogue: add bias, scatter b-major rows -> time-major gi[w*B + b]
  const int lane = tid & 31;
  const int nn = lane & 15, mb = (lane >> 4) << 3;
  const int ncol = nblk + n0 + nn;
  const float bias = bih[ncol];
#pragma unroll
  for (int r = 0; r < 8; ++r) {
    int row = r0 + m0 + mb + r;
    int b = row >> 11, w = row & 2047;
    gi[((size_t)w * B + b) * C3 + ncol] = acc[r] + bias;
  }
}

// ---------------------------------------------------------------------------
// K10: sequential GRU scan. One persistent workgroup (16 waves). Per step:
//   gh = h @ whh^T  -> 48 WMMA 16x16 tiles (M=16 is exactly the batch!),
//   whh^T bf16 stays L2-resident; next step's gi prefetched.
// ---------------------------------------------------------------------------
__global__ __launch_bounds__(512) void gru_seq_kernel(const float* __restrict__ gi,
                                                      const bf16* __restrict__ whh_bf,
                                                      const float* __restrict__ bhh,
                                                      float* __restrict__ h_last) {
  __shared__ __align__(16) bf16 hb[16 * 256];   // h in bf16 (WMMA A operand)
  __shared__ float hf[16 * 256];                // h in f32 (state)
  __shared__ bf16 gh[16 * 768];                 // h @ whh^T result
  __shared__ float sbhh[768];
  const int tid = threadIdx.x;                  // 512 threads = 16 waves
  const int wv = tid >> 5;
  for (int i = tid; i < 768; i += 512) sbhh[i] = bhh[i];
  for (int i = tid; i < 4096; i += 512) { hf[i] = 0.f; hb[i] = (bf16)0.f; }
  __syncthreads();

  for (int t = 0; t < W; ++t) {
    const float* git = gi + (size_t)t * B * C3;
    if (t + 1 < W)
      __builtin_prefetch(gi + (size_t)(t + 1) * B * C3 + (size_t)tid * 24, 0, 0);
#pragma unroll
    for (int j = 0; j < 3; ++j) {               // 16 waves x 3 = 48 n-tiles
      const int n0 = (wv * 3 + j) << 4;
      v8f acc = {};
#pragma unroll
      for (int k0 = 0; k0 < 256; k0 += 32)
        acc = wmma_bf16(load_frag(hb + k0, 256),
                        load_frag(whh_bf + (size_t)n0 * 256 + k0, 256), acc);
      store_c_bf16(gh + n0, 768, acc);
    }
    __syncthreads();
#pragma unroll
    for (int u = 0; u < 8; ++u) {               // 4096 states / 512 threads
      int idx = tid * 8 + u;
      int b = idx >> 8, hi = idx & 255;
      const float* g = git + (size_t)b * C3;
      float ir = g[hi], iz = g[256 + hi], inn = g[512 + hi];
      float hr = (float)gh[b * 768 + hi] + sbhh[hi];
      float hz = (float)gh[b * 768 + 256 + hi] + sbhh[256 + hi];
      float hn = (float)gh[b * 768 + 512 + hi] + sbhh[512 + hi];
      float r = 1.f / (1.f + __expf(-(ir + hr)));
      float z = 1.f / (1.f + __expf(-(iz + hz)));
      float n = tanhf(inn + r * hn);
      float hnew = (1.f - z) * n + z * hf[idx];
      hf[idx] = hnew; hb[idx] = (bf16)hnew;
    }
    __syncthreads();
  }
  for (int i = tid; i < 4096; i += 512) h_last[i] = hf[i];
}

// ---------------------------------------------------------------------------
// K11: FC head + VAE (tiny: ~10 MFLOP). One block; thread n owns column n.
// ---------------------------------------------------------------------------
struct HeadP {
  const float *hl;
  const float *bn0g,*bn0b,*bn0m,*bn0v;
  const float *fc1w,*fc1b,*fc2w,*fc2b,*fc3w,*fc3b;
  const float *e1w,*e1b,*bne1g,*bne1b,*bne1m,*bne1v;
  const float *e2w,*e2b,*bne2g,*bne2b,*bne2m,*bne2v;
  const float *muw,*mub,*lvw,*lvb;
  const float *d1w,*d1b,*bnd1g,*bnd1b,*bnd1m,*bnd1v;
  const float *d2w,*d2b,*bnd2g,*bnd2b,*bnd2m,*bnd2v;
  const float *recw,*recb,*eps;
  float *out;
};

DEV void lin_layer(const float* sin, int kdim, const float* w, const float* bias,
                   float* sout, int ncols) {
  const int n = threadIdx.x;
  if (n < ncols) {
    const float* wr = w + (size_t)n * kdim;
    float acc[16];
#pragma unroll
    for (int b = 0; b < 16; ++b) acc[b] = 0.f;
    for (int k = 0; k < kdim; ++k) {
      float wv = wr[k];
#pragma unroll
      for (int b = 0; b < 16; ++b) acc[b] += sin[b * kdim + k] * wv;
    }
#pragma unroll
    for (int b = 0; b < 16; ++b) sout[b * ncols + n] = acc[b] + bias[n];
  }
}
DEV void bn_relu_col(float* s, int ncols, const float* g, const float* bb,
                     const float* m, const float* v) {
  const int n = threadIdx.x;
  if (n < ncols) {
    float sc = g[n] * rsqrtf(v[n] + 1e-5f);
    float sh = bb[n] - m[n] * sc;
#pragma unroll
    for (int b = 0; b < 16; ++b) {
      float x = s[b * ncols + n] * sc + sh;
      s[b * ncols + n] = x > 0.f ? x : 0.f;
    }
  }
}
DEV void relu_col(float* s) {
  const int n = threadIdx.x;
#pragma unroll
  for (int b = 0; b < 16; ++b) {
    float x = s[b * 256 + n];
    s[b * 256 + n] = x > 0.f ? x : 0.f;
  }
}

__global__ __launch_bounds__(256) void head_kernel(HeadP p) {
  __shared__ float s0[16 * 256], s1[16 * 256], s2[16 * 256];
  __shared__ float smu[16 * 64], slv[16 * 64], sz[16 * 64];
  const int n = threadIdx.x;
  for (int i = n; i < 4096; i += 256) s0[i] = p.hl[i];
  __syncthreads();
  // fc head: relu -> bn0 -> fc1/relu -> fc2/relu -> fc3/sigmoid
  {
    float sc = p.bn0g[n] * rsqrtf(p.bn0v[n] + 1e-5f);
    float sh = p.bn0b[n] - p.bn0m[n] * sc;
#pragma unroll
    for (int b = 0; b < 16; ++b) {
      float v = s0[b * 256 + n]; v = v > 0.f ? v : 0.f;
      s1[b * 256 + n] = v * sc + sh;
    }
  }
  __syncthreads();
  lin_layer(s1, 256, p.fc1w, p.fc1b, s2, 256); __syncthreads();
  relu_col(s2);                                __syncthreads();
  lin_layer(s2, 256, p.fc2w, p.fc2b, s1, 256); __syncthreads();
  relu_col(s1);                                __syncthreads();
  lin_layer(s1, 256, p.fc3w, p.fc3b, s2, 256); __syncthreads();
#pragma unroll
  for (int b = 0; b < 16; ++b)
    p.out[b * 256 + n] = 1.f / (1.f + __expf(-s2[b * 256 + n]));
  __syncthreads();
  // VAE encoder
  lin_layer(s0, 256, p.e1w, p.e1b, s1, 256);   __syncthreads();
  bn_relu_col(s1, 256, p.bne1g, p.bne1b, p.bne1m, p.bne1v); __syncthreads();
  lin_layer(s1, 256, p.e2w, p.e2b, s2, 256);   __syncthreads();
  bn_relu_col(s2, 256, p.bne2g, p.bne2b, p.bne2m, p.bne2v); __syncthreads();
  lin_layer(s2, 256, p.muw, p.mub, smu, 64);   __syncthreads();
  lin_layer(s2, 256, p.lvw, p.lvb, slv, 64);   __syncthreads();
  if (n < 64) {
#pragma unroll
    for (int b = 0; b < 16; ++b) {
      float m = smu[b * 64 + n], l = slv[b * 64 + n];
      p.out[8192 + b * 64 + n] = m;
      p.out[9216 + b * 64 + n] = l;
      sz[b * 64 + n] = m + p.eps[n] * __expf(0.5f * l);
    }
  }
  __syncthreads();
  // VAE decoder
  lin_layer(sz, 64, p.d1w, p.d1b, s1, 256);    __syncthreads();
  bn_relu_col(s1, 256, p.bnd1g, p.bnd1b, p.bnd1m, p.bnd1v); __syncthreads();
  lin_layer(s1, 256, p.d2w, p.d2b, s2, 256);   __syncthreads();
  bn_relu_col(s2, 256, p.bnd2g, p.bnd2b, p.bnd2m, p.bnd2v); __syncthreads();
  lin_layer(s2, 256, p.recw, p.recb, s1, 256); __syncthreads();
#pragma unroll
  for (int b = 0; b < 16; ++b)
    p.out[4096 + b * 256 + n] = 1.f / (1.f + __expf(-s1[b * 256 + n]));
}

// ---------------------------------------------------------------------------
extern "C" void kernel_launch(void* const* d_in, const int* in_sizes, int n_in,
                              void* d_out, int out_size, void* d_ws, size_t ws_size,
                              hipStream_t stream) {
  (void)in_sizes; (void)n_in; (void)out_size; (void)ws_size;
  const float* x    = (const float*)d_in[0];
  const float* wk_t = (const float*)d_in[1];
  const float* wq_t = (const float*)d_in[2];
  const float* wk_f = (const float*)d_in[3];
  const float* wq_f = (const float*)d_in[4];
  const float* wih  = (const float*)d_in[5];
  const float* whh  = (const float*)d_in[6];
  const float* bih  = (const float*)d_in[7];
  const float* bhh  = (const float*)d_in[8];
  float* out = (float*)d_out;

  char* ws = (char*)d_ws;
  size_t off = 0;
  auto alloc = [&](size_t bytes) -> void* {
    void* p = ws + off;
    off = (off + bytes + 255) & ~(size_t)255;
    return p;
  };
  float* pooled  = (float*)alloc((size_t)B * D * W * 4);       // 33.5 MB
  bf16*  cat     = (bf16*) alloc((size_t)B * W * C3 * 2);      // 50.3 MB
  float* gi      = (float*)alloc((size_t)W * B * C3 * 4);      // 100.7 MB
  float* kt      = (float*)alloc((size_t)B * W * 4);
  float* qt      = (float*)alloc((size_t)B * W * 4);
  float* rmaxt   = (float*)alloc((size_t)B * W * 4);
  float* rsclt   = (float*)alloc((size_t)B * W * 4);
  float* qmaxt   = (float*)alloc(B * 4);
  float* kf      = (float*)alloc((size_t)B * D * 4);
  float* qf      = (float*)alloc((size_t)B * D * 4);
  float* qmaxf   = (float*)alloc(B * 4);
  bf16*  attnbf  = (bf16*) alloc((size_t)B * D * D * 2);       // 2.1 MB
  bf16*  whhbf   = (bf16*) alloc((size_t)C3 * 256 * 2);
  float* hlast   = (float*)alloc((size_t)B * 256 * 4);

  // 1) prep: conv+maxpool, xp -> cat[:, :, 0:256]
  prep_kernel<<<B * (W / 16), 256, 0, stream>>>(x, pooled, cat);
  // 2) time GAT scalars + softmax stats
  time_kq_kernel<<<B * W / 256, 256, 0, stream>>>(pooled, wk_t, wq_t, kt, qt);
  colmax_kernel<<<B, 256, 0, stream>>>(qt, W, qmaxt);
  rowstats_kernel<<<B * W, 256, 0, stream>>>(kt, qt, qmaxt, W, W, rmaxt, rsclt);
  // 3) fused softmax x value WMMA GEMM -> cat[:, :, 512:768]
  gat_time_gemm<<<B * (W / 16), 256, 0, stream>>>(pooled, kt, qt, rmaxt, rsclt, cat);
  // 4) feature GAT: scalars, attention matrix (output!), WMMA GEMM
  feat_kq_kernel<<<B * D / 256, 256, 0, stream>>>(pooled, wk_f, wq_f, kf, qf);
  colmax_kernel<<<B, 256, 0, stream>>>(qf, D, qmaxf);
  feat_attn_kernel<<<B * D, 256, 0, stream>>>(kf, qf, qmaxf, out + 10240, attnbf);
  gat_feat_gemm<<<B * (W / 16), 256, 0, stream>>>(pooled, attnbf, cat);
  // 5) GRU: batched input projection (WMMA), then sequential scan (WMMA M=16)
  f32_to_bf16_kernel<<<C3 * 256 / 256, 256, 0, stream>>>(whh, whhbf, C3 * 256);
  gru_in_gemm<<<(B * W / 32) * 12, 256, 0, stream>>>(cat, wih, bih, gi);
  gru_seq_kernel<<<1, 512, 0, stream>>>(gi, whhbf, bhh, hlast);
  // 6) FC + VAE heads
  HeadP p;
  p.hl = hlast;
  p.bn0g=(const float*)d_in[9];  p.bn0b=(const float*)d_in[10];
  p.bn0m=(const float*)d_in[11]; p.bn0v=(const float*)d_in[12];
  p.fc1w=(const float*)d_in[13]; p.fc1b=(const float*)d_in[14];
  p.fc2w=(const float*)d_in[15]; p.fc2b=(const float*)d_in[16];
  p.fc3w=(const float*)d_in[17]; p.fc3b=(const float*)d_in[18];
  p.e1w=(const float*)d_in[19];  p.e1b=(const float*)d_in[20];
  p.bne1g=(const float*)d_in[21]; p.bne1b=(const float*)d_in[22];
  p.bne1m=(const float*)d_in[23]; p.bne1v=(const float*)d_in[24];
  p.e2w=(const float*)d_in[25];  p.e2b=(const float*)d_in[26];
  p.bne2g=(const float*)d_in[27]; p.bne2b=(const float*)d_in[28];
  p.bne2m=(const float*)d_in[29]; p.bne2v=(const float*)d_in[30];
  p.muw=(const float*)d_in[31];  p.mub=(const float*)d_in[32];
  p.lvw=(const float*)d_in[33];  p.lvb=(const float*)d_in[34];
  p.d1w=(const float*)d_in[35];  p.d1b=(const float*)d_in[36];
  p.bnd1g=(const float*)d_in[37]; p.bnd1b=(const float*)d_in[38];
  p.bnd1m=(const float*)d_in[39]; p.bnd1v=(const float*)d_in[40];
  p.d2w=(const float*)d_in[41];  p.d2b=(const float*)d_in[42];
  p.bnd2g=(const float*)d_in[43]; p.bnd2b=(const float*)d_in[44];
  p.bnd2m=(const float*)d_in[45]; p.bnd2v=(const float*)d_in[46];
  p.recw=(const float*)d_in[47]; p.recb=(const float*)d_in[48];
  p.eps=(const float*)d_in[49];
  p.out = out;
  head_kernel<<<1, 256, 0, stream>>>(p);
}
